// MultiHeadAttention_62423054680262
// MI455X (gfx1250) — matrix operand order repaired and checked by reading, not compiled, not run on hardware
//
#include <hip/hip_runtime.h>
#include <math.h>

// ---------------------------------------------------------------------------
// MultiHeadAttention for MI455X (gfx1250), wave32, f16 WMMA w/ f32 accumulate.
// Pipeline:
//   pack x,w_qkv -> f16 WMMA fragments
//   GEMM1: qkv = x @ w_qkv + b_qkv          (v_wmma_f32_16x16x32_f16)
//   pack  Q (A-frags), K^T (B-frags), V (B-frags) per head
//   flash attention (online softmax, no 1GB score matrix)
//   pack attn, w_out -> fragments
//   GEMM2: out = attn @ w_out + b_out
// GEMM inner loop is 2-stage software pipelined (ping-pong fragment regs).
// ---------------------------------------------------------------------------

typedef __attribute__((ext_vector_type(16))) _Float16 v16h;
typedef __attribute__((ext_vector_type(8)))  _Float16 v8h;
typedef __attribute__((ext_vector_type(8)))  float    v8f;
typedef __attribute__((ext_vector_type(4)))  float    v4f;

#define S_LEN 4096
#define DM    1024
#define NH    16
#define DK    64

#define WMMA_F16(a, b, c) \
  __builtin_amdgcn_wmma_f32_16x16x32_f16(false, (a), false, (b), (short)0, (c), false, false)

// ---------------------------------------------------------------------------
// Fragment packing.  A-fragment (16x32, MxK), per CDNA5 ISA:
//   lane L (L<16): row M=L,  K = 0..7 and 16..23
//   lane L+16    : row M=L,  K = 8..15 and 24..31
// B-fragment (32x16, KxN):
//   lanes 0-15 : col N=lane,     K = 0..7 and 16..23 (2 per VGPR)
//   lanes 16-31: col N=lane-16,  K = 8..15 and 24..31
// Tile storage: 512 halves per tile, lane slot = tile*512 + lane*16.
// ---------------------------------------------------------------------------

__global__ void __launch_bounds__(256)
pack_a_f32(const float* __restrict__ src, int ld, int colOff, int colStridePerHead,
           int Kt, int tilesTotal, _Float16* __restrict__ dst, int dstTilesPerHead) {
  int h    = blockIdx.y;
  int t    = blockIdx.x * blockDim.x + threadIdx.x;
  int tile = t >> 5, lane = t & 31;
  if (tile >= tilesTotal) return;
  int mt = tile / Kt, kt = tile % Kt;
  const float* s   = src + (size_t)(colOff + h * colStridePerHead);
  int m  = mt * 16 + (lane & 15);
  int kb = kt * 32 + ((lane >> 4) << 3);
  const float* row = s + (size_t)m * ld;
  // 8 consecutive floats at kb, 8 more at kb+16 -> four 16B loads (aligned)
  v4f c0 = *(const v4f*)(row + kb);
  v4f c1 = *(const v4f*)(row + kb + 4);
  v4f c2 = *(const v4f*)(row + kb + 16);
  v4f c3 = *(const v4f*)(row + kb + 20);
  v16h out;
#pragma unroll
  for (int i = 0; i < 4; ++i) {
    out[i]      = (_Float16)c0[i];
    out[4 + i]  = (_Float16)c1[i];
    out[8 + i]  = (_Float16)c2[i];
    out[12 + i] = (_Float16)c3[i];
  }
  ((v16h*)(dst + (size_t)h * dstTilesPerHead * 512))[tile * 32 + lane] = out;
}

// B from row-major K'xN source: B[kk][n] = src[kk*ld + colOff + n]
__global__ void __launch_bounds__(256)
pack_b_f32(const float* __restrict__ src, int ld, int colOff, int colStridePerHead,
           int Kt, int tilesTotal, _Float16* __restrict__ dst, int dstTilesPerHead) {
  int h    = blockIdx.y;
  int t    = blockIdx.x * blockDim.x + threadIdx.x;
  int tile = t >> 5, lane = t & 31;
  if (tile >= tilesTotal) return;
  int nt = tile / Kt, kt = tile % Kt;
  const float* s = src + (size_t)(colOff + h * colStridePerHead);
  int n  = nt * 16 + (lane & 15);
  int k0 = kt * 32 + ((lane >> 4) << 3);
  v16h out;
#pragma unroll
  for (int i = 0; i < 16; ++i) out[i] = (_Float16)s[(size_t)(k0 + (i & 7) + ((i >> 3) << 4)) * ld + n];
  ((v16h*)(dst + (size_t)h * dstTilesPerHead * 512))[tile * 32 + lane] = out;
}

// B from transposed source: B[kk][n] = src[n*ld + colOff + kk]   (builds K^T frags)
__global__ void __launch_bounds__(256)
pack_bT_f32(const float* __restrict__ src, int ld, int colOff, int colStridePerHead,
            int Kt, int tilesTotal, _Float16* __restrict__ dst, int dstTilesPerHead) {
  int h    = blockIdx.y;
  int t    = blockIdx.x * blockDim.x + threadIdx.x;
  int tile = t >> 5, lane = t & 31;
  if (tile >= tilesTotal) return;
  int nt = tile / Kt, kt = tile % Kt;
  const float* s = src + (size_t)(colOff + h * colStridePerHead);
  int n  = nt * 16 + (lane & 15);
  int k0 = kt * 32 + ((lane >> 4) << 3);
  const float* row = s + (size_t)n * ld;
  v4f c0 = *(const v4f*)(row + k0);
  v4f c1 = *(const v4f*)(row + k0 + 4);
  v4f c2 = *(const v4f*)(row + k0 + 16);
  v4f c3 = *(const v4f*)(row + k0 + 20);
  v16h out;
#pragma unroll
  for (int i = 0; i < 4; ++i) {
    out[i]      = (_Float16)c0[i];
    out[4 + i]  = (_Float16)c1[i];
    out[8 + i]  = (_Float16)c2[i];
    out[12 + i] = (_Float16)c3[i];
  }
  ((v16h*)(dst + (size_t)h * dstTilesPerHead * 512))[tile * 32 + lane] = out;
}

// ---------------------------------------------------------------------------
// Fragment GEMM: C[M,N] = A@B + bias.  8 waves/block, wave = 64(M) x 32(N).
// grid = (N/256, M/64).  8 WMMA per K-step, 6 fragment loads, 2-stage pipeline.
// C/D layout: lane l holds col n=l&15, rows r+8*(l>>4).
// Kt must be even (both call sites use Kt=32).
// ---------------------------------------------------------------------------
__global__ void __launch_bounds__(256)
gemm_f16frag(const _Float16* __restrict__ A, const _Float16* __restrict__ B,
             const float* __restrict__ bias, float* __restrict__ C,
             int Kt, int ldc) {
  int lane = threadIdx.x & 31;
  int wave = threadIdx.x >> 5;
  int nt0  = (blockIdx.x * 8 + wave) * 2;
  int mt0  = blockIdx.y * 4;
  const v16h* Afl = (const v16h*)A + lane;
  const v16h* Bfl = (const v16h*)B + lane;

  v8f acc[8];
#pragma unroll
  for (int i = 0; i < 8; ++i) acc[i] = (v8f){};

  v16h a0[4], a1[4], b0[2], b1[2];
#pragma unroll
  for (int i = 0; i < 4; ++i) a0[i] = Afl[(size_t)((mt0 + i) * Kt) * 32];
#pragma unroll
  for (int j = 0; j < 2; ++j) b0[j] = Bfl[(size_t)((nt0 + j) * Kt) * 32];

  for (int kt = 0; kt < Kt; kt += 2) {
    // stage 1 prefetch (kt+1)
#pragma unroll
    for (int i = 0; i < 4; ++i) a1[i] = Afl[(size_t)((mt0 + i) * Kt + kt + 1) * 32];
#pragma unroll
    for (int j = 0; j < 2; ++j) b1[j] = Bfl[(size_t)((nt0 + j) * Kt + kt + 1) * 32];
    // consume stage 0 (kt)
#pragma unroll
    for (int i = 0; i < 4; ++i) {
      acc[2 * i]     = WMMA_F16(a0[i], b0[0], acc[2 * i]);
      acc[2 * i + 1] = WMMA_F16(a0[i], b0[1], acc[2 * i + 1]);
    }
    // stage 0 prefetch (kt+2)
    if (kt + 2 < Kt) {
#pragma unroll
      for (int i = 0; i < 4; ++i) a0[i] = Afl[(size_t)((mt0 + i) * Kt + kt + 2) * 32];
#pragma unroll
      for (int j = 0; j < 2; ++j) b0[j] = Bfl[(size_t)((nt0 + j) * Kt + kt + 2) * 32];
    }
    // consume stage 1 (kt+1)
#pragma unroll
    for (int i = 0; i < 4; ++i) {
      acc[2 * i]     = WMMA_F16(a1[i], b1[0], acc[2 * i]);
      acc[2 * i + 1] = WMMA_F16(a1[i], b1[1], acc[2 * i + 1]);
    }
  }

  int n0   = nt0 * 16 + (lane & 15);
  float bv0 = bias[n0];
  float bv1 = bias[n0 + 16];
  int row0 = mt0 * 16 + ((lane >> 4) << 3);
#pragma unroll
  for (int i = 0; i < 4; ++i) {
#pragma unroll
    for (int r = 0; r < 8; ++r) {
      float* dst = C + (size_t)(row0 + i * 16 + r) * ldc;
      dst[n0]      = acc[2 * i][r] + bv0;
      dst[n0 + 16] = acc[2 * i + 1][r] + bv1;
    }
  }
}

// ---------------------------------------------------------------------------
// Flash attention: one wave per (head, 16-query tile).  Keys in blocks of 32.
// All 8 fragment loads (4 K^T + 4 V) are hoisted to the top of the iteration
// so the softmax VALU/TRANS section hides the V-fragment latency.
// S = Q K^T * 0.125 via 4 WMMA; online softmax (row stats via shfl_xor within
// 16-lane halves, matching C-layout row striping); P reshaped C->A via a
// per-wave LDS bounce; O += P V via 4 WMMA.  attn output: [4096][1024] f32.
// ---------------------------------------------------------------------------
__global__ void __launch_bounds__(256)
flash_attn(const _Float16* __restrict__ Aq, const _Float16* __restrict__ Bk,
           const _Float16* __restrict__ Bv, float* __restrict__ attn) {
  __shared__ __align__(16) _Float16 lds_p[8][16][32];   // 8 KB, per-wave slabs
  int lane = threadIdx.x & 31;
  int widx = threadIdx.x >> 5;
  int gw   = blockIdx.x * 8 + widx;
  int h    = gw >> 8;         // 256 query tiles per head
  int qt   = gw & 255;

  const v16h* AqF = (const v16h*)(Aq + (size_t)h * S_LEN * DK) + lane;
  const v16h* BkF = (const v16h*)(Bk + (size_t)h * S_LEN * DK) + lane;
  const v16h* BvF = (const v16h*)(Bv + (size_t)h * S_LEN * DK) + lane;

  v16h q0 = AqF[(size_t)(qt * 2 + 0) * 32];
  v16h q1 = AqF[(size_t)(qt * 2 + 1) * 32];

  v8f o0 = {}, o1 = {}, o2 = {}, o3 = {};
  float mrun[8], lrun[8];
#pragma unroll
  for (int r = 0; r < 8; ++r) { mrun[r] = -1e30f; lrun[r] = 0.0f; }
  const float scale = 0.125f;   // 1/sqrt(64)

  for (int kb = 0; kb < S_LEN / 32; ++kb) {
    int kt0 = 2 * kb, kt1 = 2 * kb + 1;          // two 16-key column tiles
    // hoisted fragment loads for this key block
    v16h bk0 = BkF[(size_t)(kt0 * 2 + 0) * 32];
    v16h bk1 = BkF[(size_t)(kt0 * 2 + 1) * 32];
    v16h bk2 = BkF[(size_t)(kt1 * 2 + 0) * 32];
    v16h bk3 = BkF[(size_t)(kt1 * 2 + 1) * 32];
    v16h bv0 = BvF[(size_t)(0 * 128 + kb) * 32];
    v16h bv1 = BvF[(size_t)(1 * 128 + kb) * 32];
    v16h bv2 = BvF[(size_t)(2 * 128 + kb) * 32];
    v16h bv3 = BvF[(size_t)(3 * 128 + kb) * 32];

    v8f s0 = {}, s1 = {};
    s0 = WMMA_F16(q0, bk0, s0);
    s0 = WMMA_F16(q1, bk1, s0);
    s1 = WMMA_F16(q0, bk2, s1);
    s1 = WMMA_F16(q1, bk3, s1);

#pragma unroll
    for (int r = 0; r < 8; ++r) {
      float x0 = s0[r] * scale, x1 = s1[r] * scale;
      float v = fmaxf(x0, x1);
      v = fmaxf(v, __shfl_xor(v, 1));
      v = fmaxf(v, __shfl_xor(v, 2));
      v = fmaxf(v, __shfl_xor(v, 4));
      v = fmaxf(v, __shfl_xor(v, 8));
      float mnew = fmaxf(mrun[r], v);
      float p0 = __expf(x0 - mnew);
      float p1 = __expf(x1 - mnew);
      float rs = p0 + p1;
      rs += __shfl_xor(rs, 1);
      rs += __shfl_xor(rs, 2);
      rs += __shfl_xor(rs, 4);
      rs += __shfl_xor(rs, 8);
      float alpha = __expf(mrun[r] - mnew);
      lrun[r] = lrun[r] * alpha + rs;
      mrun[r] = mnew;
      o0[r] *= alpha; o1[r] *= alpha; o2[r] *= alpha; o3[r] *= alpha;
      int m  = r + ((lane >> 4) << 3);
      int nl = lane & 15;
      lds_p[widx][m][nl]      = (_Float16)p0;
      lds_p[widx][m][16 + nl] = (_Float16)p1;
    }
    asm volatile("s_wait_dscnt 0" ::: "memory");   // same-wave DS store->load
    int m     = lane & 15;
    int kbase = ((lane >> 4) << 3);
    v8h lo = *(const v8h*)&lds_p[widx][m][kbase];
    v8h hi = *(const v8h*)&lds_p[widx][m][16 + kbase];
    v16h p;
#pragma unroll
    for (int i = 0; i < 8; ++i) { p[i] = lo[i]; p[8 + i] = hi[i]; }

    o0 = WMMA_F16(p, bv0, o0);
    o1 = WMMA_F16(p, bv1, o1);
    o2 = WMMA_F16(p, bv2, o2);
    o3 = WMMA_F16(p, bv3, o3);
  }

  int row0 = qt * 16 + ((lane >> 4) << 3);
  int col0 = h * DK + (lane & 15);
#pragma unroll
  for (int r = 0; r < 8; ++r) {
    float inv = 1.0f / lrun[r];
    float* dst = attn + (size_t)(row0 + r) * DM + col0;
    dst[0]  = o0[r] * inv;
    dst[16] = o1[r] * inv;
    dst[32] = o2[r] * inv;
    dst[48] = o3[r] * inv;
  }
}

// ---------------------------------------------------------------------------
// Host-side orchestration (graph-capture safe: only launches on `stream`).
// Workspace layout (bytes), peak ~109 MB:
//   [0      ,  8 MB) A_x frags      -> reused for A_o frags
//   [8 MB   , 14 MB) B_wqkv frags   -> reused for B_wout frags
//   [16 MB  , 64 MB) qkv f32 [4096][3072]
//   [64 MB  , 72 MB) A_q frags (per head)
//   [72 MB  , 80 MB) B_kT frags (per head)
//   [80 MB  , 88 MB) B_v frags (per head)
//   [88 MB  ,104 MB) attn f32 [4096][1024]
// ---------------------------------------------------------------------------
extern "C" void kernel_launch(void* const* d_in, const int* in_sizes, int n_in,
                              void* d_out, int out_size, void* d_ws, size_t ws_size,
                              hipStream_t stream) {
  (void)in_sizes; (void)n_in; (void)out_size; (void)ws_size;
  const float* x     = (const float*)d_in[0];
  const float* w_qkv = (const float*)d_in[1];
  const float* b_qkv = (const float*)d_in[2];
  const float* w_out = (const float*)d_in[3];
  const float* b_out = (const float*)d_in[4];
  float* out = (float*)d_out;

  char* ws = (char*)d_ws;
  const size_t MB = 1024 * 1024;
  _Float16* Ax   = (_Float16*)(ws + 0);
  _Float16* Bw1  = (_Float16*)(ws + 8 * MB);
  float*    qkvf = (float*)   (ws + 16 * MB);
  _Float16* Aq   = (_Float16*)(ws + 64 * MB);
  _Float16* Bk   = (_Float16*)(ws + 72 * MB);
  _Float16* Bv   = (_Float16*)(ws + 80 * MB);
  float*    attnf= (float*)   (ws + 88 * MB);
  _Float16* Ao   = (_Float16*)(ws + 0);       // reuse (Ax dead after GEMM1)
  _Float16* Bw2  = (_Float16*)(ws + 8 * MB);  // reuse (Bw1 dead after GEMM1)

  // 1) x [4096,1024] -> A frags (Mt=256, Kt=32, 8192 tiles)
  pack_a_f32<<<dim3(1024, 1), 256, 0, stream>>>(x, DM, 0, 0, 32, 8192, Ax, 0);
  // 2) w_qkv [1024,3072] -> B frags (Kt=32, Nt=192, 6144 tiles)
  pack_b_f32<<<dim3(768, 1), 256, 0, stream>>>(w_qkv, 3 * DM, 0, 0, 32, 6144, Bw1, 0);
  // 3) qkv = x @ w_qkv + b_qkv   (M=4096, N=3072, K=1024); block = 64M x 256N
  gemm_f16frag<<<dim3(12, 64), 256, 0, stream>>>(Ax, Bw1, b_qkv, qkvf, 32, 3 * DM);
  // 4) Q per head -> A frags (Mt=256, Kt=2, 512 tiles/head)
  pack_a_f32<<<dim3(64, NH), 256, 0, stream>>>(qkvf, 3 * DM, 0, DK, 2, 512, Aq, 512);
  // 5) K^T per head -> B frags (Kt=2 over d, Nt=256 over keys)
  pack_bT_f32<<<dim3(64, NH), 256, 0, stream>>>(qkvf, 3 * DM, DM, DK, 2, 512, Bk, 512);
  // 6) V per head -> B frags (Kt=128 over keys, Nt=4 over d)
  pack_b_f32<<<dim3(64, NH), 256, 0, stream>>>(qkvf, 3 * DM, 2 * DM, DK, 128, 512, Bv, 512);
  // 7) flash attention: 16 heads x 256 q-tiles = 4096 waves
  flash_attn<<<dim3(512, 1), 256, 0, stream>>>(Aq, Bk, Bv, attnf);
  // 8) attn [4096,1024] -> A frags
  pack_a_f32<<<dim3(1024, 1), 256, 0, stream>>>(attnf, DM, 0, 0, 32, 8192, Ao, 0);
  // 9) w_out [1024,1024] -> B frags (Kt=32, Nt=64, 2048 tiles)
  pack_b_f32<<<dim3(256, 1), 256, 0, stream>>>(w_out, DM, 0, 0, 32, 2048, Bw2, 0);
  // 10) out = attn @ w_out + b_out   (M=4096, N=1024, K=1024)
  gemm_f16frag<<<dim3(4, 64), 256, 0, stream>>>(Ao, Bw2, b_out, out, 32, DM);
}
